// GCNConvNetBN_41180146434792
// MI455X (gfx1250) — matrix operand order repaired and checked by reading, not compiled, not run on hardware
//
#include <hip/hip_runtime.h>
#include <hip/hip_bf16.h>

typedef __bf16 v16bf __attribute__((ext_vector_type(16)));
typedef __bf16 v8bf  __attribute__((ext_vector_type(8)));
typedef float  v8f   __attribute__((ext_vector_type(8)));

#define NNODES 100000
#define NEDGES 200000
#define NGRAPHS 1000
#define NODES_PER_GRAPH 100
#define MAXCH 1024

// ---------------------------------------------------------------- utility
__global__ void zero_f32(float* __restrict__ p, size_t n) {
    size_t i = (size_t)blockIdx.x * blockDim.x + threadIdx.x;
    if (i < n) p[i] = 0.0f;
}

// ---------------------------------------------------------------- degree / norms
__global__ void count_deg(const int* __restrict__ dst, float* __restrict__ deg, int E) {
    int e = blockIdx.x * blockDim.x + threadIdx.x;
    if (e < E) atomicAdd(&deg[dst[e]], 1.0f);
}
__global__ void node_norm(const float* __restrict__ deg, float* __restrict__ dis,
                          float* __restrict__ self_norm, int N) {
    int n = blockIdx.x * blockDim.x + threadIdx.x;
    if (n < N) {
        float d = 1.0f + deg[n];
        float r = rsqrtf(d);
        dis[n] = r;
        self_norm[n] = 1.0f / d;
    }
}
__global__ void edge_norm(const int* __restrict__ src, const int* __restrict__ dst,
                          const float* __restrict__ dis, float* __restrict__ enorm, int E) {
    int e = blockIdx.x * blockDim.x + threadIdx.x;
    if (e < E) enorm[e] = dis[src[e]] * dis[dst[e]];
}

// transpose weights:  Wt[n][k] = W[k][n]   (tiny; keeps GEMM B-staging K-contiguous)
__global__ void transpose_w(const float* __restrict__ W, float* __restrict__ Wt,
                            int cin, int cout) {
    int idx = blockIdx.x * blockDim.x + threadIdx.x;
    if (idx >= cin * cout) return;
    int k = idx / cout, n = idx % cout;
    Wt[(size_t)n * cin + k] = W[idx];
}

// ---------------------------------------------------------------- WMMA GEMM
// XW = bn(X) @ W.  bn folds previous layer's BatchNorm: x*scale[c] + shift[c]
// (identity when scale == nullptr).  64x64 block tile, K-step 32, 8 waves.
// LDS tiles are stored in WMMA *fragment order*: per (tile, lane) a contiguous
// 32-byte v16bf region -> fragment loads are 2x ds_load_b128 per operand.
#define BM 64
#define BN 64
#define BK 32

__global__ __launch_bounds__(256) void gemm_bf16_wmma(
    const float* __restrict__ X, const float* __restrict__ scale,
    const float* __restrict__ shift, const float* __restrict__ Wt,  // [cout][cin]
    float* __restrict__ XW, int N, int cin, int cout)
{
    // fragment-ordered LDS: [tile][lane][16 bf16]
    __shared__ alignas(32) __bf16 AsF[4][32][16];   // 4 KB
    __shared__ alignas(32) __bf16 BsF[4][32][16];   // 4 KB

    const int bm   = blockIdx.x * BM;
    const int bn   = blockIdx.y * BN;
    const int tid  = threadIdx.x;
    const int wave = tid >> 5;
    const int lane = tid & 31;

    // staging job: one K-octet (8 values) per thread, 256 jobs = full A or B tile
    const int jrow = tid >> 2;        // 0..63  (A: row m in tile; B: col n in tile)
    const int joct = tid & 3;         // 0..3   K-octet
    // A fragment placement (ISA: lanes 0-15 K{0-7,16-23}, lanes 16-31 K{8-15,24-31})
    const int a_tile = jrow >> 4;
    const int a_lane = (jrow & 15) + 16 * (joct & 1);
    const int a_half = joct >> 1;
    // B fragment placement (same as A: lanes 0-15 K{0-7,16-23}, lanes 16-31 K{8-15,24-31})
    const int b_tile = jrow >> 4;
    const int b_lane = (jrow & 15) + 16 * (joct & 1);
    const int b_half = joct >> 1;

    const int gmA = bm + jrow;                    // global row for A staging
    const int gnB = bn + jrow;                    // global col for B staging
    const bool vec_ok = ((cin & 3) == 0);         // 16B-aligned octet loads possible

    const int mt = (wave & 3);                    // wave's m-tile
    const int nt = (wave >> 2) * 2;               // wave's first n-tile

    v8f acc0 = {};
    v8f acc1 = {};

    for (int k0 = 0; k0 < cin; k0 += BK) {
        const int kk = k0 + joct * 8;             // first K of this thread's octet
        // ---- fetch A octet (fp32, BN folded) into registers
        float av[8];
        if (gmA < N && vec_ok && kk + 8 <= cin) {
            const float4 lo = *(const float4*)(X + (size_t)gmA * cin + kk);
            const float4 hi = *(const float4*)(X + (size_t)gmA * cin + kk + 4);
            av[0] = lo.x; av[1] = lo.y; av[2] = lo.z; av[3] = lo.w;
            av[4] = hi.x; av[5] = hi.y; av[6] = hi.z; av[7] = hi.w;
        } else {
#pragma unroll
            for (int j = 0; j < 8; ++j)
                av[j] = (gmA < N && kk + j < cin) ? X[(size_t)gmA * cin + kk + j] : 0.0f;
        }
        if (scale) {
#pragma unroll
            for (int j = 0; j < 8; ++j)
                if (kk + j < cin) av[j] = av[j] * scale[kk + j] + shift[kk + j];
        }
        // ---- fetch B octet (Wt is [cout][cin] so K is contiguous)
        float bv[8];
        if (gnB < cout && vec_ok && kk + 8 <= cin) {
            const float4 lo = *(const float4*)(Wt + (size_t)gnB * cin + kk);
            const float4 hi = *(const float4*)(Wt + (size_t)gnB * cin + kk + 4);
            bv[0] = lo.x; bv[1] = lo.y; bv[2] = lo.z; bv[3] = lo.w;
            bv[4] = hi.x; bv[5] = hi.y; bv[6] = hi.z; bv[7] = hi.w;
        } else {
#pragma unroll
            for (int j = 0; j < 8; ++j)
                bv[j] = (gnB < cout && kk + j < cin) ? Wt[(size_t)gnB * cin + kk + j] : 0.0f;
        }

        __syncthreads();   // previous iteration's fragment reads complete
        {
            v8bf a8, b8;
#pragma unroll
            for (int j = 0; j < 8; ++j) { a8[j] = (__bf16)av[j]; b8[j] = (__bf16)bv[j]; }
            *(v8bf*)&AsF[a_tile][a_lane][a_half * 8] = a8;   // ds_store_b128
            *(v8bf*)&BsF[b_tile][b_lane][b_half * 8] = b8;   // ds_store_b128
        }
        __syncthreads();   // tiles visible

        const v16bf a  = *(const v16bf*)AsF[mt][lane];        // 2x ds_load_b128
        const v16bf b0 = *(const v16bf*)BsF[nt][lane];
        const v16bf b1 = *(const v16bf*)BsF[nt + 1][lane];

        acc0 = __builtin_amdgcn_wmma_f32_16x16x32_bf16(false, a, false, b0,
                                                       (short)0, acc0, false, false);
        acc1 = __builtin_amdgcn_wmma_f32_16x16x32_bf16(false, a, false, b1,
                                                       (short)0, acc1, false, false);
    }

    // C/D layout: VGPR r: lanes 0-15 -> M=r, lanes 16-31 -> M=r+8; N=lane&15
    const int nloc   = lane & 15;
    const int m_half = (lane >> 4) * 8;
#pragma unroll
    for (int r = 0; r < 8; ++r) {
        int gm  = bm + mt * 16 + m_half + r;
        int gn0 = bn + nt * 16 + nloc;
        int gn1 = bn + (nt + 1) * 16 + nloc;
        if (gm < N) {
            if (gn0 < cout) XW[(size_t)gm * cout + gn0] = acc0[r];
            if (gn1 < cout) XW[(size_t)gm * cout + gn1] = acc1[r];
        }
    }
}

// ---------------------------------------------------------------- aggregation
// out[n,c] = xw[n,c] * self_norm[n] + bias[c]
__global__ void init_out(const float* __restrict__ xw, const float* __restrict__ self_norm,
                         const float* __restrict__ bias, float* __restrict__ out,
                         int N, int cout) {
    size_t idx = (size_t)blockIdx.x * blockDim.x + threadIdx.x;
    if (idx >= (size_t)N * cout) return;
    int n = (int)(idx / cout);
    int c = (int)(idx % cout);
    out[idx] = xw[idx] * self_norm[n] + bias[c];
}

// out[dst[e], c] += xw[src[e], c] * enorm[e]  (4 channels / thread, float4 gather)
__global__ void scatter_edges(const float* __restrict__ xw, const int* __restrict__ src,
                              const int* __restrict__ dst, const float* __restrict__ enorm,
                              float* __restrict__ out, int E, int cout) {
    int cg = cout >> 2;                      // all couts divisible by 4
    size_t idx = (size_t)blockIdx.x * blockDim.x + threadIdx.x;
    if (idx >= (size_t)E * cg) return;
    int e  = (int)(idx / cg);
    int c4 = (int)(idx % cg) * 4;
    float en = enorm[e];
    const float4 v = *(const float4*)(xw + (size_t)src[e] * cout + c4);
    float* o = out + (size_t)dst[e] * cout + c4;
    atomicAdd(o + 0, v.x * en);
    atomicAdd(o + 1, v.y * en);
    atomicAdd(o + 2, v.z * en);
    atomicAdd(o + 3, v.w * en);
}

// fused ReLU (in place) + per-channel sum / sumsq reduction
__global__ __launch_bounds__(256) void relu_stats(float* __restrict__ out,
                                                  float* __restrict__ stats,
                                                  int N, int cout, int rowsPerBlock) {
    int r0 = blockIdx.x * rowsPerBlock;
    int r1 = r0 + rowsPerBlock; if (r1 > N) r1 = N;
    for (int c = threadIdx.x; c < cout; c += 256) {
        float s = 0.0f, sq = 0.0f;
        for (int r = r0; r < r1; ++r) {
            size_t i = (size_t)r * cout + c;
            float v = out[i];
            v = v > 0.0f ? v : 0.0f;
            out[i] = v;
            s += v; sq += v * v;
        }
        atomicAdd(&stats[c], s);
        atomicAdd(&stats[cout + c], sq);
    }
}

// fold BN into per-channel affine for the consumer
__global__ void make_scaleshift(const float* __restrict__ stats, const float* __restrict__ gamma,
                                const float* __restrict__ beta, float* __restrict__ scale,
                                float* __restrict__ shift, int N, int cout) {
    int c = blockIdx.x * blockDim.x + threadIdx.x;
    if (c >= cout) return;
    float mean = stats[c] / (float)N;
    float var  = stats[cout + c] / (float)N - mean * mean;
    float sc   = gamma[c] * rsqrtf(var + 1e-5f);
    scale[c] = sc;
    shift[c] = beta[c] - mean * sc;
}

// ---------------------------------------------------------------- pool + FC
__global__ __launch_bounds__(64) void pool_fc(const float* __restrict__ h,
                                              const float* __restrict__ scale,
                                              const float* __restrict__ shift,
                                              const float* __restrict__ fc_w,
                                              const float* __restrict__ fc_b,
                                              float* __restrict__ out,
                                              int cout, int nclass) {
    __shared__ float emb[64];
    int g = blockIdx.x;
    int t = threadIdx.x;
    if (t < cout) {
        float sc = scale[t], sh = shift[t];
        const float* base = h + (size_t)g * NODES_PER_GRAPH * cout + t;
        float m = -3.4e38f;
        for (int r = 0; r < NODES_PER_GRAPH; ++r) {
            float v = base[(size_t)r * cout] * sc + sh;
            m = fmaxf(m, v);
        }
        emb[t] = m;
    }
    __syncthreads();
    if (t < nclass) {
        float s = fc_b[t];
        for (int c = 0; c < cout; ++c) s += emb[c] * fc_w[t * cout + c];
        out[g * nclass + t] = s;
    }
}

// ---------------------------------------------------------------- host orchestration
extern "C" void kernel_launch(void* const* d_in, const int* in_sizes, int n_in,
                              void* d_out, int out_size, void* d_ws, size_t ws_size,
                              hipStream_t stream) {
    (void)in_sizes; (void)n_in; (void)out_size; (void)ws_size;

    const int N = NNODES, E = NEDGES;
    const float* x   = (const float*)d_in[0];
    const int*   ei  = (const int*)d_in[1];
    const int*   src = ei;
    const int*   dst = ei + E;
    const float* Ws[8], *Bi[8], *Ga[8], *Be[8];
    for (int i = 0; i < 8; ++i) {
        Ws[i] = (const float*)d_in[3 + i];
        Bi[i] = (const float*)d_in[11 + i];
        Ga[i] = (const float*)d_in[19 + i];
        Be[i] = (const float*)d_in[27 + i];
    }
    const float* fc_w = (const float*)d_in[35];
    const float* fc_b = (const float*)d_in[36];

    // carve workspace
    char* p = (char*)d_ws;
    auto alloc = [&](size_t bytes) -> float* {
        float* r = (float*)p;
        p += (bytes + 255) & ~(size_t)255;
        return r;
    };
    float* deg       = alloc((size_t)N * 4);
    float* dis       = alloc((size_t)N * 4);
    float* self_norm = alloc((size_t)N * 4);
    float* enorm     = alloc((size_t)E * 4);
    float* stats     = alloc((size_t)2 * MAXCH * 4);
    float* scale     = alloc((size_t)MAXCH * 4);
    float* shift     = alloc((size_t)MAXCH * 4);
    float* wt        = alloc((size_t)MAXCH * MAXCH * 4);   // transposed weights
    const size_t actBytes = (size_t)N * MAXCH * 4;
    float* bufs[3] = { alloc(actBytes), alloc(actBytes), alloc(actBytes) };

    // degree + symmetric GCN norms
    zero_f32<<<(N + 255) / 256, 256, 0, stream>>>(deg, (size_t)N);
    count_deg<<<(E + 255) / 256, 256, 0, stream>>>(dst, deg, E);
    node_norm<<<(N + 255) / 256, 256, 0, stream>>>(deg, dis, self_norm, N);
    edge_norm<<<(E + 255) / 256, 256, 0, stream>>>(src, dst, dis, enorm, E);

    const int ch[9] = {3, 64, 64, 64, 128, 1024, 512, 256, 40};
    const float* act = x;
    const float* curScale = nullptr;
    const float* curShift = nullptr;

    for (int i = 0; i < 8; ++i) {
        const int cin = ch[i], cout = ch[i + 1];
        float* xw  = bufs[(2 * i) % 3];
        float* out = bufs[(2 * i + 1) % 3];

        transpose_w<<<(cin * cout + 255) / 256, 256, 0, stream>>>(Ws[i], wt, cin, cout);

        dim3 grid((N + BM - 1) / BM, (cout + BN - 1) / BN);
        gemm_bf16_wmma<<<grid, 256, 0, stream>>>(act, curScale, curShift, wt,
                                                 xw, N, cin, cout);

        size_t tot = (size_t)N * cout;
        init_out<<<(unsigned)((tot + 255) / 256), 256, 0, stream>>>(
            xw, self_norm, Bi[i], out, N, cout);

        size_t etot = (size_t)E * (cout / 4);
        scatter_edges<<<(unsigned)((etot + 255) / 256), 256, 0, stream>>>(
            xw, src, dst, enorm, out, E, cout);

        zero_f32<<<(2 * cout + 255) / 256, 256, 0, stream>>>(stats, (size_t)2 * cout);
        const int rpb = 128;
        relu_stats<<<(N + rpb - 1) / rpb, 256, 0, stream>>>(out, stats, N, cout, rpb);
        make_scaleshift<<<(cout + 255) / 256, 256, 0, stream>>>(
            stats, Ga[i], Be[i], scale, shift, N, cout);

        act = out;
        curScale = scale;
        curShift = shift;
    }

    pool_fc<<<NGRAPHS, 64, 0, stream>>>(act, scale, shift, fc_w, fc_b,
                                        (float*)d_out, ch[8], 2);
}
